// BaseFeatureExtraction_37855841747571
// MI455X (gfx1250) — matrix operand order repaired and checked by reading, not compiled, not run on hardware
//
#include <hip/hip_runtime.h>
#include <math.h>

// ---------------- constants ----------------
#define NB    8
#define HDIM  256
#define WDIM  256
#define HW    65536          // HDIM*WDIM
#define NW    4              // waves per workgroup in the WMMA conv
#define NT    4              // 16-pixel N-tiles per wave (A-fragment reuse)

typedef _Float16 half_t;
typedef __attribute__((ext_vector_type(16))) _Float16 v16h;
typedef __attribute__((ext_vector_type(8)))  _Float16 v8h;
typedef __attribute__((ext_vector_type(8)))  float    v8f;

// =====================================================================
// Weight repack: f32 OIHW -> f16 in WMMA A-fragment order.
// packed[((ct*nCh + c)*32 + lane)*16 + j] =
//   w[co = ct*16 + lane%16][k = c*32 + j + 16*(lane/16)]
// K ordered (kh,kw,ci) so B-fragments are channel-contiguous in NHWC.
// =====================================================================
__global__ void k_repack(const float* __restrict__ w, half_t* __restrict__ out,
                         int Cin, int Cout, int KS, int total)
{
    int id = blockIdx.x * blockDim.x + threadIdx.x;
    if (id >= total) return;
    int j    = id & 15;
    int lane = (id >> 4) & 31;
    int r    = id >> 9;              // ct*nCh + c
    int Ktot = Cin * KS * KS;
    int nCh  = Ktot >> 5;
    int c    = r % nCh;
    int ct   = r / nCh;
    int co   = ct * 16 + (lane & 15);
    int g    = lane >> 4;
    int k    = c * 32 + j + (g << 4);
    int flat;
    if (KS == 1) {
        flat = co * Cin + k;
    } else {
        int tap = k / Cin, ci = k - tap * Cin;
        int kh = tap / 3, kw = tap - kh * 3;
        flat = ((co * Cin + ci) * 3 + kh) * 3 + kw;
    }
    out[id] = (half_t)w[flat];
}

// ---------------------------------------------------------------------
// shared epilogue: bias / residual / relu, aligned v8h store
// ---------------------------------------------------------------------
__device__ __forceinline__
void conv_epilogue(v8f* acc, const int* gp, int g, int co0,
                   const float* bias, const half_t* resid, int resC,
                   half_t* out, int outC, int relu)
{
    const int mB = co0 + (g << 3);              // 8 contiguous channels
    #pragma unroll
    for (int t = 0; t < NT; ++t) {
        size_t n = (size_t)gp[t];
        v8h rv = {};
        if (resid) rv = *(const v8h*)(resid + n * resC + mB);
        v8h o;
        #pragma unroll
        for (int r = 0; r < 8; ++r) {
            float v = acc[t][r];
            if (bias)  v += bias[mB + r];
            if (resid) v += (float)rv[r];
            if (relu)  v  = fmaxf(v, 0.0f);
            o[r] = (half_t)v;
        }
        *(v8h*)(out + n * outC + mB) = o;
    }
}

// =====================================================================
// 1x1 conv: implicit GEMM, f16 NHWC, packed f16 weights.
// One v16h B-load + one WMMA per (chunk, tile); A reused across NT tiles.
// =====================================================================
__global__ __launch_bounds__(32 * NW)
void k_wmma_conv1(const half_t* __restrict__ in, int inC,
                  const half_t* __restrict__ wp,
                  const float*  __restrict__ bias,
                  const half_t* __restrict__ resid, int resC,
                  half_t* __restrict__ out, int outC,
                  int Cin, int relu)
{
    const int lane = threadIdx.x & 31;
    const int wave = threadIdx.x >> 5;
    const int g    = lane >> 4;
    const int l16  = lane & 15;
    const int nCh  = Cin >> 5;

    const int pixBase = blockIdx.x * (16 * NT * NW) + wave * (16 * NT);
    int gp[NT];
    #pragma unroll
    for (int t = 0; t < NT; ++t) gp[t] = pixBase + t * 16 + l16;

    const half_t* __restrict__ wbase =
        wp + ((size_t)blockIdx.y * nCh * 32 + lane) * 16;

    v8f acc[NT] = {};
    for (int c = 0; c < nCh; ++c) {
        v16h a = *(const v16h*)(wbase + (size_t)c * 512);
        int k0c = c * 32 + (g << 4);
        #pragma unroll
        for (int t = 0; t < NT; ++t) {
            v16h bm = *(const v16h*)(in + (size_t)gp[t] * inC + k0c);
            acc[t] = __builtin_amdgcn_wmma_f32_16x16x32_f16(
                         false, a, false, bm, (short)0, acc[t], false, false);
        }
    }
    conv_epilogue(acc, gp, g, blockIdx.y * 16, bias, resid, resC, out, outC, relu);
}

// =====================================================================
// 3x3 conv (PAD: 0 = zero pad, 1 = reflect), tap-outer structure:
// clamp + address math done once per (tap, tile); inner loop over the
// Cin/32 channel chunks of that tap is pure load -> wmma.
// =====================================================================
template<int PAD>
__global__ __launch_bounds__(32 * NW)
void k_wmma_conv3(const half_t* __restrict__ in, int inC,
                  const half_t* __restrict__ wp,
                  const float*  __restrict__ bias,
                  const half_t* __restrict__ resid, int resC,
                  half_t* __restrict__ out, int outC,
                  int Cin, int relu)
{
    const int lane = threadIdx.x & 31;
    const int wave = threadIdx.x >> 5;
    const int g    = lane >> 4;
    const int l16  = lane & 15;
    const int nPT  = Cin >> 5;                 // chunks per tap

    const int pixBase = blockIdx.x * (16 * NT * NW) + wave * (16 * NT);
    const int bb      = pixBase >> 16;         // batch, uniform
    int gp[NT], hh[NT], wv[NT];
    #pragma unroll
    for (int t = 0; t < NT; ++t) {
        gp[t] = pixBase + t * 16 + l16;
        int p = gp[t] & (HW - 1);
        hh[t] = p >> 8;
        wv[t] = p & 255;
    }
    const size_t batchOff = (size_t)bb * HW;
    const half_t* __restrict__ wbase =
        wp + ((size_t)blockIdx.y * (nPT * 9) * 32 + lane) * 16;

    v8f acc[NT] = {};
    #pragma unroll
    for (int kh = -1; kh <= 1; ++kh) {
        #pragma unroll
        for (int kw = -1; kw <= 1; ++kw) {
            const int tap = (kh + 1) * 3 + (kw + 1);
            size_t off[NT];
            bool   ok[NT];
            #pragma unroll
            for (int t = 0; t < NT; ++t) {
                int sh = hh[t] + kh, sw = wv[t] + kw;
                if (PAD) {          // reflect: always in-bounds
                    sh = (sh < 0) ? -sh : ((sh >= HDIM) ? 2 * HDIM - 2 - sh : sh);
                    sw = (sw < 0) ? -sw : ((sw >= WDIM) ? 2 * WDIM - 2 - sw : sw);
                    ok[t] = true;
                } else {            // zero pad: mask, clamp address
                    ok[t] = (sh >= 0) & (sh < HDIM) & (sw >= 0) & (sw < WDIM);
                    sh = min(max(sh, 0), HDIM - 1);
                    sw = min(max(sw, 0), WDIM - 1);
                }
                off[t] = (batchOff + (size_t)(sh * WDIM + sw)) * inC + (g << 4);
            }
            const half_t* wtap = wbase + (size_t)tap * nPT * 512;
            for (int cc = 0; cc < nPT; ++cc) {
                v16h a = *(const v16h*)(wtap + (size_t)cc * 512);
                #pragma unroll
                for (int t = 0; t < NT; ++t) {
                    v16h bm = {};
                    if (PAD || ok[t])
                        bm = *(const v16h*)(in + off[t] + cc * 32);
                    acc[t] = __builtin_amdgcn_wmma_f32_16x16x32_f16(
                                 false, a, false, bm, (short)0, acc[t], false, false);
                }
            }
        }
    }
    conv_epilogue(acc, gp, g, blockIdx.y * 16, bias, resid, resC, out, outC, relu);
}

// =====================================================================
// pre conv: 1x1 3->32 + bias + relu.  f32 NCHW in, f16 NHWC out.
// =====================================================================
__global__ void k_pre1(const float* __restrict__ x, const float* __restrict__ w,
                       const float* __restrict__ bi, half_t* __restrict__ out)
{
    int gp = blockIdx.x * blockDim.x + threadIdx.x;   // b*HW + p
    int p = gp & (HW - 1);
    int b = gp >> 16;
    const float* xi = x + (size_t)b * 3 * HW + p;
    float i0 = xi[0], i1 = xi[HW], i2 = xi[2 * HW];
    half_t* o = out + (size_t)gp * 32;
    #pragma unroll
    for (int co = 0; co < 32; ++co) {
        float v = bi[co] + w[co * 3] * i0 + w[co * 3 + 1] * i1 + w[co * 3 + 2] * i2;
        o[co] = (half_t)fmaxf(v, 0.0f);
    }
}

// in-place layernorm over 64 channels (contiguous in NHWC)
__global__ void k_ln64(half_t* __restrict__ buf,
                       const float* __restrict__ w, const float* __restrict__ bi)
{
    int gp = blockIdx.x * blockDim.x + threadIdx.x;
    half_t* base = buf + (size_t)gp * 64;
    float v[64], s = 0.0f, s2 = 0.0f;
    #pragma unroll
    for (int c = 0; c < 64; ++c) {
        v[c] = (float)base[c];
        s += v[c]; s2 += v[c] * v[c];
    }
    float mu  = s * (1.0f / 64.0f);
    float var = s2 * (1.0f / 64.0f) - mu * mu;
    float inv = rsqrtf(var + 1e-5f);
    #pragma unroll
    for (int c = 0; c < 64; ++c)
        base[c] = (half_t)((v[c] - mu) * inv * w[c] + bi[c]);
}

// =====================================================================
// fused attention statistics: one block per (b, head).
// streams q,k once; 8x8 gram + q/k sq-norms in registers; shfl reduce;
// softmax(scale * q^ k^T) -> attn[(b*4+h)*64 + i*8 + j]
// =====================================================================
__global__ void k_attn(const half_t* __restrict__ qkv, const float* __restrict__ scale,
                       float* __restrict__ attn)
{
    __shared__ float swave[8 * 80];
    __shared__ float res[80];
    int bh = blockIdx.x, b = bh >> 2, h = bh & 3;
    int lane = threadIdx.x & 31, wave = threadIdx.x >> 5;

    float pacc[64], qs[8], ks[8];
    #pragma unroll
    for (int i = 0; i < 64; ++i) pacc[i] = 0.0f;
    #pragma unroll
    for (int i = 0; i < 8; ++i) { qs[i] = 0.0f; ks[i] = 0.0f; }

    for (int n = threadIdx.x; n < HW; n += 256) {
        const half_t* base = qkv + ((size_t)b * HW + n) * 96;
        v8h qv = *(const v8h*)(base + h * 8);
        v8h kv = *(const v8h*)(base + 32 + h * 8);
        float q[8], k[8];
        #pragma unroll
        for (int i = 0; i < 8; ++i) { q[i] = (float)qv[i]; k[i] = (float)kv[i]; }
        #pragma unroll
        for (int i = 0; i < 8; ++i) {
            qs[i] += q[i] * q[i];
            ks[i] += k[i] * k[i];
            #pragma unroll
            for (int j = 0; j < 8; ++j) pacc[i * 8 + j] += q[i] * k[j];
        }
    }
    #pragma unroll
    for (int v = 0; v < 64; ++v) {
        float x = pacc[v];
        #pragma unroll
        for (int o = 16; o > 0; o >>= 1) x += __shfl_xor(x, o, 32);
        if (lane == 0) swave[wave * 80 + v] = x;
    }
    #pragma unroll
    for (int v = 0; v < 8; ++v) {
        float x = qs[v];
        #pragma unroll
        for (int o = 16; o > 0; o >>= 1) x += __shfl_xor(x, o, 32);
        if (lane == 0) swave[wave * 80 + 64 + v] = x;
        float y = ks[v];
        #pragma unroll
        for (int o = 16; o > 0; o >>= 1) y += __shfl_xor(y, o, 32);
        if (lane == 0) swave[wave * 80 + 72 + v] = y;
    }
    __syncthreads();
    if (threadIdx.x < 80) {
        float s = 0.0f;
        #pragma unroll
        for (int w = 0; w < 8; ++w) s += swave[w * 80 + threadIdx.x];
        res[threadIdx.x] = s;
    }
    __syncthreads();
    if (threadIdx.x == 0) {
        float sc = scale[h];
        float qn[8], kn[8];
        #pragma unroll
        for (int i = 0; i < 8; ++i) {
            qn[i] = fmaxf(sqrtf(res[64 + i]), 1e-12f);
            kn[i] = fmaxf(sqrtf(res[72 + i]), 1e-12f);
        }
        #pragma unroll
        for (int i = 0; i < 8; ++i) {
            float L[8], m = -1e30f;
            #pragma unroll
            for (int j = 0; j < 8; ++j) {
                L[j] = res[i * 8 + j] / (qn[i] * kn[j]) * sc;
                m = fmaxf(m, L[j]);
            }
            float s = 0.0f;
            #pragma unroll
            for (int j = 0; j < 8; ++j) { L[j] = __expf(L[j] - m); s += L[j]; }
            float inv = 1.0f / s;
            #pragma unroll
            for (int j = 0; j < 8; ++j) attn[bh * 64 + i * 8 + j] = L[j] * inv;
        }
    }
}

// out[b][n][c] = sum_d attn[b][h][c%8][d] * v[b][n][64 + h*8 + d]
__global__ void k_attn_out(const half_t* __restrict__ qkv, const float* __restrict__ attn,
                           half_t* __restrict__ out)
{
    int gp = blockIdx.x * blockDim.x + threadIdx.x;   // b*HW + p
    int b = gp >> 16;
    const half_t* vb = qkv + (size_t)gp * 96 + 64;
    float v[32];
    #pragma unroll
    for (int d = 0; d < 32; ++d) v[d] = (float)vb[d];
    const float* ab = attn + b * 256;
    half_t* o = out + (size_t)gp * 32;
    #pragma unroll
    for (int c = 0; c < 32; ++c) {
        int h = c >> 3, i = c & 7;
        const float* ar = ab + h * 64 + i * 8;
        float acc = 0.0f;
        #pragma unroll
        for (int d = 0; d < 8; ++d) acc += ar[d] * v[h * 8 + d];
        o[c] = (half_t)acc;
    }
}

// lightweight conv: per-channel 1D conv (k=3, zero pad) over flattened HW.
// input = channels 32..63 of RES (NHWC rowC=64); 32 outputs per thread.
__global__ void k_lconv(const half_t* __restrict__ in, const float* __restrict__ lc,
                        half_t* __restrict__ out)
{
    int gp = blockIdx.x * blockDim.x + threadIdx.x;
    int p  = gp & (HW - 1);
    const half_t* xm = in + (size_t)gp * 64 + 32;
    half_t* o = out + (size_t)gp * 32;
    bool hasL = (p > 0), hasR = (p < HW - 1);
    #pragma unroll
    for (int c = 0; c < 32; ++c) {
        const float* wl = lc + (c & 3) * 3;
        float acc = wl[1] * (float)xm[c];
        if (hasL) acc += wl[0] * (float)xm[c - 64];
        if (hasR) acc += wl[2] * (float)xm[c + 64];
        o[c] = (half_t)acc;
    }
}

// fused grouped 3x3 dwconv (128ch, groups=64) + gelu(h1)*h2 gate -> 64ch
__global__ void k_dwgate(const half_t* __restrict__ m1, const float* __restrict__ wd,
                         half_t* __restrict__ out)
{
    int idx = blockIdx.x * blockDim.x + threadIdx.x;  // over NB*HW*64
    int co = idx & 63;
    int gp = idx >> 6;
    int p  = gp & (HW - 1);
    int b  = gp >> 16;
    int h = p >> 8, w = p & 255;
    int ci0 = co & ~1;
    const half_t* base = m1 + (size_t)b * HW * 128;
    const float* w1 = wd + co * 18;
    const float* w2 = wd + (co + 64) * 18;
    float a1 = 0.0f, a2 = 0.0f;
    #pragma unroll
    for (int kh = 0; kh < 3; ++kh) {
        int sh = h + kh - 1;
        if (sh < 0 || sh >= HDIM) continue;
        #pragma unroll
        for (int kw = 0; kw < 3; ++kw) {
            int sw = w + kw - 1;
            if (sw < 0 || sw >= WDIM) continue;
            const half_t* px = base + (size_t)(sh * WDIM + sw) * 128;
            int t = kh * 3 + kw;
            a1 += w1[t] * (float)px[ci0]      + w1[9 + t] * (float)px[ci0 + 1];
            a2 += w2[t] * (float)px[ci0 + 64] + w2[9 + t] * (float)px[ci0 + 65];
        }
    }
    float g = 0.5f * a1 * (1.0f + erff(a1 * 0.70710678118654752f));
    out[(size_t)gp * 64 + co] = (half_t)(g * a2);
}

// post conv: 1x1 32->3 + bias (f16 NHWC -> f16 NHWC, 3ch)
__global__ void k_post2(const half_t* __restrict__ in, const float* __restrict__ w,
                        const float* __restrict__ bi, half_t* __restrict__ out)
{
    int gp = blockIdx.x * blockDim.x + threadIdx.x;
    const half_t* xi = in + (size_t)gp * 32;
    float v[32];
    #pragma unroll
    for (int ci = 0; ci < 32; ++ci) v[ci] = (float)xi[ci];
    #pragma unroll
    for (int co = 0; co < 3; ++co) {
        float acc = bi[co];
        #pragma unroll
        for (int ci = 0; ci < 32; ++ci) acc += w[co * 32 + ci] * v[ci];
        out[(size_t)gp * 3 + co] = (half_t)acc;
    }
}

// post conv: 3x3 3->3, reflect pad, bias, relu.  f16 NHWC -> f32 NCHW d_out
__global__ void k_post3(const half_t* __restrict__ in, const float* __restrict__ w,
                        const float* __restrict__ bi, float* __restrict__ out)
{
    int gp = blockIdx.x * blockDim.x + threadIdx.x;
    int p = gp & (HW - 1);
    int b = gp >> 16;
    int h = p >> 8, ww = p & 255;
    const half_t* xb = in + (size_t)b * HW * 3;
    float acc[3];
    #pragma unroll
    for (int co = 0; co < 3; ++co) acc[co] = bi[co];
    #pragma unroll
    for (int kh = 0; kh < 3; ++kh) {
        int sh = h + kh - 1;
        sh = (sh < 0) ? -sh : ((sh >= HDIM) ? 2 * HDIM - 2 - sh : sh);
        #pragma unroll
        for (int kw = 0; kw < 3; ++kw) {
            int sw = ww + kw - 1;
            sw = (sw < 0) ? -sw : ((sw >= WDIM) ? 2 * WDIM - 2 - sw : sw);
            const half_t* px = xb + (size_t)(sh * WDIM + sw) * 3;
            #pragma unroll
            for (int ci = 0; ci < 3; ++ci) {
                float xv = (float)px[ci];
                #pragma unroll
                for (int co = 0; co < 3; ++co)
                    acc[co] += w[((co * 3 + ci) * 3 + kh) * 3 + kw] * xv;
            }
        }
    }
    #pragma unroll
    for (int co = 0; co < 3; ++co)
        out[((size_t)b * 3 + co) * HW + p] = fmaxf(acc[co], 0.0f);
}

// =====================================================================
// launch
// =====================================================================
extern "C" void kernel_launch(void* const* d_in, const int* in_sizes, int n_in,
                              void* d_out, int out_size, void* d_ws, size_t ws_size,
                              hipStream_t stream)
{
    const float* x       = (const float*)d_in[0];
    const float* pre_w1  = (const float*)d_in[1];
    const float* pre_b1  = (const float*)d_in[2];
    const float* pre_w2  = (const float*)d_in[3];
    const float* pre_b2  = (const float*)d_in[4];
    const float* pre_w3  = (const float*)d_in[5];
    const float* pre_b3  = (const float*)d_in[6];
    const float* n1_w    = (const float*)d_in[7];
    const float* n1_b    = (const float*)d_in[8];
    const float* ascale  = (const float*)d_in[9];
    const float* qkv1_w  = (const float*)d_in[10];
    const float* qkv2_w  = (const float*)d_in[11];
    const float* proj_w  = (const float*)d_in[12];
    const float* lc_w    = (const float*)d_in[13];
    const float* p1_w    = (const float*)d_in[14];
    const float* lnc_w   = (const float*)d_in[15];
    const float* lnc_b   = (const float*)d_in[16];
    const float* p2_w    = (const float*)d_in[17];
    const float* n2_w    = (const float*)d_in[18];
    const float* n2_b    = (const float*)d_in[19];
    const float* mlpin_w = (const float*)d_in[20];
    const float* mlpdw_w = (const float*)d_in[21];
    const float* mlpout_w= (const float*)d_in[22];
    const float* post_w1 = (const float*)d_in[23];
    const float* post_b1 = (const float*)d_in[24];
    const float* post_w2 = (const float*)d_in[25];
    const float* post_b2 = (const float*)d_in[26];
    const float* post_w3 = (const float*)d_in[27];
    const float* post_b3 = (const float*)d_in[28];
    float* out = (float*)d_out;

    // ---- workspace (f16 NHWC activation regions, reused) ----
    const size_t NP   = (size_t)NB * HW;          // 524288 pixels
    half_t* RES = (half_t*)d_ws;                  // 64ch: t2 -> LN -> residual
    half_t* rA  = RES + NP * 64;                  // 64ch: t1 -> concat/h0 -> gate
    half_t* rB  = rA  + NP * 64;                  // 96ch qkv_a / 64ch p1 / 128ch M1
    half_t* rC  = rB  + NP * 128;                 // 96ch qkv / 64ch X / 3ch PB
    half_t* rD  = rC  + NP * 96;                  // 32ch: t0 / AO / LC / PA
    half_t* WP  = rD  + NP * 32;                  // packed f16 weights (<160K halves)
    float*  attnBuf = (float*)(WP + 160 * 1024);  // 2048 floats

    // ---- repack all WMMA conv weights into A-fragment order ----
    size_t wo = 0;
    auto repack = [&](const float* w, int Cin, int Cout, int KS) -> half_t* {
        int total = Cout * Cin * KS * KS;
        half_t* dst = WP + wo; wo += (size_t)total;
        k_repack<<<(total + 255) / 256, 256, 0, stream>>>(w, dst, Cin, Cout, KS, total);
        return dst;
    };
    half_t* W_pre2  = repack(pre_w2,  32, 64, 1);
    half_t* W_pre3  = repack(pre_w3,  64, 64, 3);
    half_t* W_qkv1  = repack(qkv1_w,  32, 96, 1);
    half_t* W_qkv2  = repack(qkv2_w,  96, 96, 3);
    half_t* W_proj  = repack(proj_w,  32, 32, 1);
    half_t* W_p1    = repack(p1_w,    32, 64, 1);
    half_t* W_p2    = repack(p2_w,    64, 32, 1);
    half_t* W_min   = repack(mlpin_w, 64, 128, 1);
    half_t* W_mout  = repack(mlpout_w,64, 64, 1);
    half_t* W_post1 = repack(post_w1, 64, 32, 1);

    const unsigned GX = (unsigned)(NP / (16 * NT * NW));   // 2048 pixel blocks

    auto conv1 = [&](const half_t* in, int inC, const half_t* wp, const float* bias,
                     const half_t* resid, int resC, half_t* o, int outC,
                     int Cin, int Cout, int relu) {
        k_wmma_conv1<<<dim3(GX, Cout / 16), 32 * NW, 0, stream>>>(
            in, inC, wp, bias, resid, resC, o, outC, Cin, relu);
    };

    const int PIX_BLOCKS = (int)(NP / 256);       // 2048

    // ---- pre stage ----
    k_pre1<<<PIX_BLOCKS, 256, 0, stream>>>(x, pre_w1, pre_b1, rD);            // t0 (32)
    conv1(rD, 32, W_pre2, pre_b2, nullptr, 0, rA, 64, 32, 64, 0);             // t1 (64)
    k_wmma_conv3<1><<<dim3(GX, 4), 32 * NW, 0, stream>>>(                     // t2 reflect+relu
        rA, 64, W_pre3, pre_b3, nullptr, 0, RES, 64, 64, 1);
    k_ln64<<<PIX_BLOCKS, 256, 0, stream>>>(RES, n1_w, n1_b);                  // residual

    // ---- attention branch (z1 = RES channels 0..31) ----
    conv1(RES, 64, W_qkv1, nullptr, nullptr, 0, rB, 96, 32, 96, 0);           // qkv_a
    k_wmma_conv3<0><<<dim3(GX, 6), 32 * NW, 0, stream>>>(                     // qkv zero-pad
        rB, 96, W_qkv2, nullptr, nullptr, 0, rC, 96, 96, 0);
    k_attn<<<NB * 4, 256, 0, stream>>>(rC, ascale, attnBuf);                  // gram+softmax
    k_attn_out<<<PIX_BLOCKS, 256, 0, stream>>>(rC, attnBuf, rD);              // AO (32)
    conv1(rD, 32, W_proj, nullptr, RES, 64, rA, 64, 32, 32, 0);               // z1+proj -> rA[:,0:32]

    // ---- lightweight-conv branch (z2 = RES channels 32..63) ----
    k_lconv<<<PIX_BLOCKS, 256, 0, stream>>>(RES, lc_w, rD);                   // LC (32)
    conv1(rD, 32, W_p1, nullptr, nullptr, 0, rB, 64, 32, 64, 0);              // p1 (rB free)
    k_ln64<<<PIX_BLOCKS, 256, 0, stream>>>(rB, lnc_w, lnc_b);
    conv1(rB, 64, W_p2, nullptr, RES + 32, 64, rA + 32, 64, 64, 32, 0);       // z2+y -> rA[:,32:64]

    // ---- MLP ----
    k_ln64<<<PIX_BLOCKS, 256, 0, stream>>>(rA, n2_w, n2_b);                   // h0 in place
    conv1(rA, 64, W_min, nullptr, nullptr, 0, rB, 128, 64, 128, 0);           // M1 (128)
    k_dwgate<<<(int)(NP * 64 / 256), 256, 0, stream>>>(rB, mlpdw_w, rA);      // G (64)
    conv1(rA, 64, W_mout, nullptr, RES, 64, rC, 64, 64, 64, 0);               // X = resid + mlp

    // ---- post stage ----
    conv1(rC, 64, W_post1, post_b1, nullptr, 0, rD, 32, 64, 32, 1);           // relu (32)
    k_post2<<<PIX_BLOCKS, 256, 0, stream>>>(rD, post_w2, post_b2, rC);        // 3ch
    k_post3<<<PIX_BLOCKS, 256, 0, stream>>>(rC, post_w3, post_b3, out);       // reflect+relu
}
